// TensorizedCircuit_1511828488805
// MI455X (gfx1250) — matrix operand order repaired and checked by reading, not compiled, not run on hardware
//
#include <hip/hip_runtime.h>

// CDNA5 / gfx1250, wave32.  WMMA f32 = f16 x f16 + f32, 16x16x32.
typedef _Float16 half8 __attribute__((ext_vector_type(8)));
typedef _Float16 v16h  __attribute__((ext_vector_type(16)));
typedef float    v8f   __attribute__((ext_vector_type(8)));

#define K_UNITS 128
#define BATCH   512

// Async global->LDS (ASYNCcnt-tracked) if the toolchain exposes it.
#if defined(__has_builtin)
#if __has_builtin(__builtin_amdgcn_global_load_async_to_lds_b128) && \
    __has_builtin(__builtin_amdgcn_s_wait_asynccnt)
#define USE_ASYNC_LDS 1
#endif
#endif
#ifndef USE_ASYNC_LDS
#define USE_ASYNC_LDS 0
#endif

#if USE_ASYNC_LDS
// Builtin signature (from hipcc diagnostic): pointee is a 16B int vector.
typedef int v4i_b128 __attribute__((vector_size(16)));
typedef __attribute__((address_space(1))) v4i_b128* async_gptr;
typedef __attribute__((address_space(3))) v4i_b128* async_lptr;
#endif

// ---------------------------------------------------------------------------
// Kernel 1: preprocess log-weights.  One block per (fold, j-row), 128 threads
// over k.  wm[f][j] = max_k W, ew[f][j][k] = exp(W - wm) as f16.
// Padded rows (j >= J, output layer) get ew = 0 so they contribute nothing.
// ---------------------------------------------------------------------------
__global__ void prep_weights(const float* __restrict__ W,
                             _Float16* __restrict__ ew,
                             float* __restrict__ wm,
                             int J, int Jpad) {
  int f = blockIdx.x / Jpad;
  int j = blockIdx.x % Jpad;
  int k = threadIdx.x;                                   // 128 threads = k
  float v = (j < J) ? W[((size_t)f * J + j) * K_UNITS + k] : -1e30f;
  float m = v;
#pragma unroll
  for (int off = 16; off; off >>= 1) m = fmaxf(m, __shfl_xor(m, off, 32));
  __shared__ float wmax[4];
  if ((threadIdx.x & 31) == 0) wmax[threadIdx.x >> 5] = m;
  __syncthreads();
  m = fmaxf(fmaxf(wmax[0], wmax[1]), fmaxf(wmax[2], wmax[3]));
  ew[((size_t)f * Jpad + j) * K_UNITS + k] =
      (j < J) ? (_Float16)__expf(v - m) : (_Float16)0.0f;
  if (k == 0) wm[(size_t)f * Jpad + j] = (j < J) ? m : 0.0f;
}

// ---------------------------------------------------------------------------
// Kernel 2: Gaussian leaf layer fused with the first fold-pair product.
// h[f][b][k] = -0.5*((x[b,2f]-mu[2f,k])^2 + (x[b,2f+1]-mu[2f+1,k])^2)
// Never materializes the 512-fold leaf tensor (saves ~134 MB of HBM traffic).
// ---------------------------------------------------------------------------
__global__ void input_product(const float* __restrict__ x,
                              const float* __restrict__ mu,
                              float* __restrict__ h) {
  int f = blockIdx.x;          // 0..255
  int b = blockIdx.y;          // 0..511
  int k = threadIdx.x;         // 0..127
  int d0 = 2 * f, d1 = 2 * f + 1;
  float x0 = x[(size_t)b * 512 + d0];
  float x1 = x[(size_t)b * 512 + d1];
  float e0 = x0 - mu[(size_t)d0 * K_UNITS + k];
  float e1 = x1 - mu[(size_t)d1 * K_UNITS + k];
  h[((size_t)f * BATCH + b) * K_UNITS + k] = -0.5f * (e0 * e0 + e1 * e1);
}

// ---------------------------------------------------------------------------
// A-fragment loader from an LDS row-major 16x128 f16 tile.
// CDNA5 16-bit A 16x32 layout: lane<16 -> row=lane, halves {K0..7, K16..23};
// lane>=16 -> row=lane-16, halves {K8..15, K24..31}.
// ---------------------------------------------------------------------------
__device__ __forceinline__ v16h load_a_frag(const _Float16* base, int lane,
                                            int koff) {
  int row = lane & 15;
  int kb  = koff + ((lane >> 4) << 3);
  half8 a0 = *(const half8*)(base + row * K_UNITS + kb);
  half8 a1 = *(const half8*)(base + row * K_UNITS + kb + 16);
  v16h r;
#pragma unroll
  for (int i = 0; i < 8; ++i) { r[i] = a0[i]; r[i + 8] = a1[i]; }
  return r;
}

// B-fragment (32x16, B[k][n] = ew[n][k]): lane<16 -> col=lane, K {0..7, 16..23};
// lane>=16 -> col=lane-16, K {8..15, 24..31}.
__device__ __forceinline__ v16h load_b_frag(const _Float16* ewRow, int lane,
                                            int koff) {
  int ks = koff + ((lane >> 4) << 3);
  half8 b0 = *(const half8*)(ewRow + ks);
  half8 b1 = *(const half8*)(ewRow + ks + 16);
  v16h r;
#pragma unroll
  for (int i = 0; i < 8; ++i) { r[i] = b0[i]; r[i + 8] = b1[i]; }
  return r;
}

// ---------------------------------------------------------------------------
// Kernel 3: fold-pair product (optional) + logsumexp sum layer via WMMA.
// Block = (fold f, 64-batch tile).  4 waves x 16 batch rows each.
// out[f][b][j] = m[b] + wm[f][j] + log( sum_k exp(g-m) * exp(W-wm) )
// The 32 KB exp-weight tile is DMA'd global->LDS asynchronously and the wait
// is deferred until after the exp/rowmax row phase (copy hidden behind VALU).
// ---------------------------------------------------------------------------
template <int NJT, bool PAIR>
__global__ void sum_layer(const float* __restrict__ hin,
                          float* __restrict__ hout,
                          const _Float16* __restrict__ ew,
                          const float* __restrict__ wm,
                          int Jout) {
  constexpr int Jpad = NJT * 16;
  int f     = blockIdx.x;
  int btile = blockIdx.y;                 // 0..7 (8 * 64 = 512 rows)
  int tid   = threadIdx.x;                // 128 threads = 4 waves
  int w     = tid >> 5;
  int lane  = tid & 31;

  extern __shared__ char smem[];
  _Float16* ewT  = (_Float16*)smem;                               // Jpad*128 f16
  _Float16* eT   = (_Float16*)(smem + Jpad * K_UNITS * 2) + w * 16 * K_UNITS;
  float*    mrow = (float*)(smem + Jpad * K_UNITS * 2 +
                            4 * 16 * K_UNITS * 2) + w * 16;

  // Stage exp-weight tile for this fold into LDS.
  {
    const _Float16* src = ew + (size_t)f * Jpad * K_UNITS;
#if USE_ASYNC_LDS
    for (int i = tid; i < Jpad * K_UNITS / 8; i += 128) {
      __builtin_amdgcn_global_load_async_to_lds_b128(
          (async_gptr)(src + (size_t)i * 8),
          (async_lptr)(ewT + (size_t)i * 8),
          0, 0);
    }
#else
    const uint4* s4 = (const uint4*)src;
    uint4* d4 = (uint4*)ewT;
    for (int i = tid; i < Jpad * K_UNITS / 8; i += 128) d4[i] = s4[i];
#endif
  }

  int rb = btile * 64 + w * 16;           // first batch row of this wave
  const float* hin0 = hin + ((size_t)(PAIR ? 2 * f : f) * BATCH) * K_UNITS;
  const float* hin1 = PAIR ? hin + ((size_t)(2 * f + 1) * BATCH) * K_UNITS
                           : hin0;

  // Per-row: pair-sum, rowmax (wave shuffle reduce), exp -> f16 tile in LDS.
  // Runs while the ew-tile DMA is in flight.
  for (int r = 0; r < 16; ++r) {
    size_t bo = (size_t)(rb + r) * K_UNITS;
    float g0 = hin0[bo + lane];
    float g1 = hin0[bo + lane + 32];
    float g2 = hin0[bo + lane + 64];
    float g3 = hin0[bo + lane + 96];
    if (PAIR) {
      g0 += hin1[bo + lane];
      g1 += hin1[bo + lane + 32];
      g2 += hin1[bo + lane + 64];
      g3 += hin1[bo + lane + 96];
    }
    float m = fmaxf(fmaxf(g0, g1), fmaxf(g2, g3));
#pragma unroll
    for (int off = 16; off; off >>= 1) m = fmaxf(m, __shfl_xor(m, off, 32));
    eT[r * K_UNITS + lane]      = (_Float16)__expf(g0 - m);
    eT[r * K_UNITS + lane + 32] = (_Float16)__expf(g1 - m);
    eT[r * K_UNITS + lane + 64] = (_Float16)__expf(g2 - m);
    eT[r * K_UNITS + lane + 96] = (_Float16)__expf(g3 - m);
    if (lane == 0) mrow[r] = m;
  }

#if USE_ASYNC_LDS
  __builtin_amdgcn_s_wait_asynccnt(0);    // ew tile resident in LDS
#endif
  __syncthreads();                        // all waves' copy portions visible

  v8f acc[NJT];
#pragma unroll
  for (int jt = 0; jt < NJT; ++jt)
#pragma unroll
    for (int i = 0; i < 8; ++i) acc[jt][i] = 0.0f;

  // 16x128 (batch x K) times 128xJpad: 4 K-chunks of 32, NJT N-tiles of 16.
#pragma unroll
  for (int kc = 0; kc < 4; ++kc) {
    v16h a = load_a_frag(eT, lane, kc * 32);
#pragma unroll
    for (int jt = 0; jt < NJT; ++jt) {
      const _Float16* ewRow = ewT + (size_t)(jt * 16 + (lane & 15)) * K_UNITS;
      v16h b = load_b_frag(ewRow, lane, kc * 32);
      acc[jt] = __builtin_amdgcn_wmma_f32_16x16x32_f16(
          false, a, false, b, (short)0, acc[jt], false, false);
    }
  }

  // Epilogue: D layout -> lane l, vgpr v holds (row = v + 8*(l>>4), col = l&15).
  const float* wmf  = wm + (size_t)f * Jpad;
  int colj  = lane & 15;
  int rbase = (lane >> 4) << 3;
#pragma unroll
  for (int jt = 0; jt < NJT; ++jt) {
    int j = jt * 16 + colj;
    float wmv = wmf[j];
#pragma unroll
    for (int v = 0; v < 8; ++v) {
      int row = rbase + v;
      float val = mrow[row] + wmv + __logf(acc[jt][v]);
      if (j < Jout)
        hout[((size_t)f * BATCH + (rb + row)) * (size_t)Jout + j] = val;
    }
  }
}

// ---------------------------------------------------------------------------
extern "C" void kernel_launch(void* const* d_in, const int* in_sizes, int n_in,
                              void* d_out, int out_size, void* d_ws,
                              size_t ws_size, hipStream_t stream) {
  const float* x  = (const float*)d_in[0];   // (512, 512, 1)
  const float* mu = (const float*)d_in[1];   // (512, 128, 1)
  // d_in[2..9] = logw for folds 256..2 ; d_in[10] = logw_out (1,10,128)
  static const int folds[8] = {256, 128, 64, 32, 16, 8, 4, 2};

  // Workspace layout (bytes): hA | hB | ew (f16) | wm (f32)
  char* ws = (char*)d_ws;
  float* hA = (float*)ws;                               // 256*512*128 f32
  size_t off = (size_t)256 * BATCH * K_UNITS * sizeof(float);
  float* hB = (float*)(ws + off);                       // 128*512*128 f32
  off += (size_t)128 * BATCH * K_UNITS * sizeof(float);
  _Float16* ewBase = (_Float16*)(ws + off);
  size_t ewOff[9], wmOff[9], e = 0, m = 0;
  for (int i = 0; i < 8; ++i) {
    ewOff[i] = e; e += (size_t)folds[i] * 128 * K_UNITS;
    wmOff[i] = m; m += (size_t)folds[i] * 128;
  }
  ewOff[8] = e; e += (size_t)16 * K_UNITS;              // padded output layer
  wmOff[8] = m; m += 16;
  off += e * sizeof(_Float16);
  off = (off + 15) & ~(size_t)15;
  float* wmBase = (float*)(ws + off);

  // 1) Weight preprocessing (9 small launches).
  for (int i = 0; i < 8; ++i)
    prep_weights<<<folds[i] * 128, 128, 0, stream>>>(
        (const float*)d_in[2 + i], ewBase + ewOff[i], wmBase + wmOff[i],
        128, 128);
  prep_weights<<<16, 128, 0, stream>>>(
      (const float*)d_in[10], ewBase + ewOff[8], wmBase + wmOff[8], 10, 16);

  // 2) Leaf layer + first pair-product -> hA (256 folds).
  input_product<<<dim3(256, BATCH), 128, 0, stream>>>(x, mu, hA);

  // 3) Level 0: sum layer in place (no pairing; each block owns its rows).
  size_t sh8 = (size_t)128 * K_UNITS * 2 + 4 * 16 * K_UNITS * 2 + 4 * 16 * 4;
  sum_layer<8, false><<<dim3(256, 8), 128, sh8, stream>>>(
      hA, hA, ewBase + ewOff[0], wmBase + wmOff[0], 128);

  // 4) Levels 1..7: pair + sum, ping-pong hA <-> hB (cross-block safety).
  float* bufs[2] = {hA, hB};
  int cur = 0;  // current data lives in bufs[cur]
  for (int i = 1; i < 8; ++i) {
    float* src = bufs[cur];
    float* dst = bufs[cur ^ 1];
    sum_layer<8, true><<<dim3(folds[i], 8), 128, sh8, stream>>>(
        src, dst, ewBase + ewOff[i], wmBase + wmOff[i], 128);
    cur ^= 1;
  }

  // 5) Final pair (2 -> 1 fold) + 10-class output mix -> d_out (512 x 1 x 10).
  size_t sh1 = (size_t)16 * K_UNITS * 2 + 4 * 16 * K_UNITS * 2 + 4 * 16 * 4;
  sum_layer<1, true><<<dim3(1, 8), 128, sh1, stream>>>(
      bufs[cur], (float*)d_out, ewBase + ewOff[8], wmBase + wmOff[8], 10);
}
